// HMMCell_26929444946010
// MI455X (gfx1250) — matrix operand order repaired and checked by reading, not compiled, not run on hardware
//
#include <hip/hip_runtime.h>

// HMM forward algorithm on gfx1250 (transposed-GEMM formulation).
// R'[j,b] = sum_i A^T[j,i] * alpha^T[i,b]  via v_wmma_f32_16x16x32_f16
// E'[j,b] = sum_c B[j,c]  * x_t[b,c]       via v_wmma_f32_16x16x4_f32
// One block (4 waves) per unit; wave owns 16 batches; lane%16 owns one batch
// column, so the state-sum reduction is lane-local + one shfl_xor(16).

typedef __attribute__((ext_vector_type(16))) _Float16 v16h;
typedef __attribute__((ext_vector_type(8)))  _Float16 v8h;
typedef __attribute__((ext_vector_type(8)))  float    v8f;
typedef __attribute__((ext_vector_type(2)))  float    v2f;

#define U_ 256
#define N_ 64
#define S_ 4
#define B_ 64
#define T_ 1024

__global__ __launch_bounds__(128) void hmm_fwd_kernel(
    const float* __restrict__ xin,   // [B, T, S]
    const float* __restrict__ tk,    // [U, N, N]
    const float* __restrict__ ek,    // [U, N, S]
    const float* __restrict__ ik,    // [U, N]
    float* __restrict__ out)         // [B, T, U]
{
  __shared__ __align__(16) _Float16 AsT[N_ * N_];       // A^T: [j][i] f16
  __shared__ float                  Bs[N_ * S_];        // softmax(emis) [j][c]
  __shared__ float                  Is[N_];             // softmax(init)
  __shared__ __align__(32) _Float16 alphaS[4][16 * N_]; // [wave][b_local][i]

  const int u    = blockIdx.x;
  const int tid  = threadIdx.x;
  const int wave = tid >> 5;
  const int lane = tid & 31;
  const int lo   = lane & 15;   // batch column (per wave)
  const int khi  = lane >> 4;   // half-wave selector
  const int b0   = wave * 16;
  const int b    = b0 + lo;     // this lane's batch

  // ---------- one-time parameter softmaxes ----------
  if (tid < N_) {
    // transition row softmax, stored TRANSPOSED: AsT[j][i] = softmax(tk[u,i,:])[j]
    const float* row = tk + ((size_t)u * N_ + tid) * N_;   // row i = tid
    float mx = -3.4e38f;
    for (int j = 0; j < N_; ++j) mx = fmaxf(mx, row[j]);
    float s = 0.f;
    for (int j = 0; j < N_; ++j) s += __expf(row[j] - mx);
    const float inv = 1.f / s;
    for (int j = 0; j < N_; ++j)
      AsT[j * N_ + tid] = (_Float16)(__expf(row[j] - mx) * inv);

    // emission row softmax (4 wide)
    const float* er = ek + ((size_t)u * N_ + tid) * S_;
    const float m2 = fmaxf(fmaxf(er[0], er[1]), fmaxf(er[2], er[3]));
    const float e0 = __expf(er[0] - m2), e1 = __expf(er[1] - m2);
    const float e2 = __expf(er[2] - m2), e3 = __expf(er[3] - m2);
    const float i2 = 1.f / (e0 + e1 + e2 + e3);
    Bs[tid * S_ + 0] = e0 * i2;
    Bs[tid * S_ + 1] = e1 * i2;
    Bs[tid * S_ + 2] = e2 * i2;
    Bs[tid * S_ + 3] = e3 * i2;
  }
  if (tid == 64) {
    const float* ir = ik + (size_t)u * N_;
    float mx = -3.4e38f;
    for (int j = 0; j < N_; ++j) mx = fmaxf(mx, ir[j]);
    float s = 0.f;
    for (int j = 0; j < N_; ++j) s += __expf(ir[j] - mx);
    const float inv = 1.f / s;
    for (int j = 0; j < N_; ++j) Is[j] = __expf(ir[j] - mx) * inv;
  }
  __syncthreads();

  // ---------- register-resident A^T as WMMA f16 A-operands ----------
  // 16-bit A-operand (16 x 32, M=j, K=i): lane%16 = row j; documented interleave
  // maps vector halves h=0..7 -> K = h + 8*khi, h=8..15 -> K = 16 + (h-8) + 8*khi.
  v16h Areg[2][4]; // [k-chunk][m-tile]
  #pragma unroll
  for (int kt = 0; kt < 2; ++kt) {
    #pragma unroll
    for (int mt = 0; mt < 4; ++mt) {
      const _Float16* src = &AsT[(mt * 16 + lo) * N_ + 8 * khi + 32 * kt];
      const v8h g0 = *(const v8h*)(src);       // K = base + 0..7
      const v8h g1 = *(const v8h*)(src + 16);  // K = base + 16..23
      Areg[kt][mt] = __builtin_shufflevector(g0, g1,
                       0, 1, 2, 3, 4, 5, 6, 7, 8, 9, 10, 11, 12, 13, 14, 15);
    }
  }

  // ---------- emission matrix as WMMA f32 16x16x4 A-operands ----------
  // 32-bit A-operand (16 x 4): lane%16 = row j; VGPR0 -> K = 2*khi, VGPR1 -> K = 2*khi+1.
  v2f Ereg[4];
  #pragma unroll
  for (int mt = 0; mt < 4; ++mt)
    Ereg[mt] = *(const v2f*)(&Bs[(mt * 16 + lo) * S_ + 2 * khi]);

  float ll = 0.f;
  const float* xptr = xin + (size_t)b * T_ * S_ + 2 * khi; // this lane's x stream
  float* outp = out + (size_t)b * T_ * U_ + u;             // this lane's ll stream

  // software-pipelined x_t (B-operand of the emission WMMA)
  v2f xv = *(const v2f*)(xptr);

  // ---------- T-step forward recurrence ----------
  for (int t = 0; t < T_; ++t) {
    // prefetch next step's observation (clamped; overlaps this step's math)
    const int tn = (t + 1 < T_) ? (t + 1) : t;
    const v2f xv_next = *(const v2f*)(xptr + (size_t)tn * S_);

    // R' in C-layout: C[mt][r], lane -> (j = mt*16 + r + 8*khi, b = lo)
    v8f C[4];
    if (t == 0) {
      #pragma unroll
      for (int mt = 0; mt < 4; ++mt)
        #pragma unroll
        for (int r = 0; r < 8; ++r)
          C[mt][r] = Is[mt * 16 + 8 * khi + r];
    } else {
      // alpha^T B-operand (32 x 16 f16, K=i, N=b): lane%16 = col b;
      // interleaved like A: h=0..7 -> K = h + 8*khi, h=8..15 -> K = 16 + (h-8) + 8*khi (+32*kt).
      v16h bop[2];
      #pragma unroll
      for (int kt = 0; kt < 2; ++kt) {
        const _Float16* src = &alphaS[wave][lo * N_ + 8 * khi + 32 * kt];
        const v8h g0 = *(const v8h*)(src);      // K = base + 0..7
        const v8h g1 = *(const v8h*)(src + 16); // K = base + 16..23
        bop[kt] = __builtin_shufflevector(g0, g1,
                    0, 1, 2, 3, 4, 5, 6, 7, 8, 9, 10, 11, 12, 13, 14, 15);
      }
      #pragma unroll
      for (int mt = 0; mt < 4; ++mt) {
        v8f z;
        #pragma unroll
        for (int i = 0; i < 8; ++i) z[i] = 0.f;
        z = __builtin_amdgcn_wmma_f32_16x16x32_f16(false, Areg[0][mt], false,
                bop[0], (short)0, z, false, false);
        z = __builtin_amdgcn_wmma_f32_16x16x32_f16(false, Areg[1][mt], false,
                bop[1], (short)0, z, false, false);
        C[mt] = z;
      }
    }

    // E' = Bsm (64x4) @ x_t^T (4x16) via f32 WMMA; B-operand: lane%16 = batch,
    // VGPR0/1 -> c = 2*khi, 2*khi+1.
    v8f E[4];
    #pragma unroll
    for (int mt = 0; mt < 4; ++mt) {
      v8f z;
      #pragma unroll
      for (int i = 0; i < 8; ++i) z[i] = 0.f;
      E[mt] = __builtin_amdgcn_wmma_f32_16x16x4_f32(false, Ereg[mt], false,
                  xv, (short)0, z, false, false);
    }

    // fwd = E .* R ; state-sum: 4 independent accumulator chains (pk-friendly),
    // then lane-local combine + one cross-half shuffle.
    float fwd[4][8];
    float a0 = 0.f, a1 = 0.f, a2 = 0.f, a3 = 0.f;
    #pragma unroll
    for (int mt = 0; mt < 4; ++mt) {
      #pragma unroll
      for (int r = 0; r < 8; ++r)
        fwd[mt][r] = E[mt][r] * C[mt][r];
      a0 += fwd[mt][0] + fwd[mt][4];
      a1 += fwd[mt][1] + fwd[mt][5];
      a2 += fwd[mt][2] + fwd[mt][6];
      a3 += fwd[mt][3] + fwd[mt][7];
    }
    float s = (a0 + a1) + (a2 + a3);
    s += __shfl_xor(s, 16, 32);

    ll += __logf(s);
    if (khi == 0)
      outp[(size_t)t * U_] = ll;

    // normalize (fast v_rcp; alpha is quantized to f16 anyway) and restage
    // alpha^T: lane's 8 rows per tile are 8 consecutive states -> one
    // ds_store_b128 per tile. Same-wave LDS ops are processed in order, so the
    // store->load RAW at the next iteration is ordered by hardware.
    const float inv = __builtin_amdgcn_rcpf(s);
    #pragma unroll
    for (int mt = 0; mt < 4; ++mt) {
      v8h a;
      #pragma unroll
      for (int r = 0; r < 8; ++r)
        a[r] = (_Float16)(fwd[mt][r] * inv);
      *(v8h*)(&alphaS[wave][lo * N_ + mt * 16 + 8 * khi]) = a;
    }

    xv = xv_next;
  }
}

extern "C" void kernel_launch(void* const* d_in, const int* in_sizes, int n_in,
                              void* d_out, int out_size, void* d_ws, size_t ws_size,
                              hipStream_t stream) {
  (void)in_sizes; (void)n_in; (void)d_ws; (void)ws_size; (void)out_size;
  const float* xin = (const float*)d_in[0]; // inputs           [64,1024,4]
  const float* tk  = (const float*)d_in[1]; // transition_kernel[256,64,64]
  const float* ek  = (const float*)d_in[2]; // emission_kernel  [256,64,4]
  const float* ik  = (const float*)d_in[3]; // init_kernel      [256,64]
  float* out = (float*)d_out;               // [64,1024,256]
  hmm_fwd_kernel<<<U_, 128, 0, stream>>>(xin, tk, ek, ik, out);
}
